// AttentionDecoder_55430847922578
// MI455X (gfx1250) — matrix operand order repaired and checked by reading, not compiled, not run on hardware
//
#include <hip/hip_runtime.h>
#include <hip/hip_bf16.h>
#include <math.h>

#define B_   128
#define T_   1024
#define E_   512
#define A_   512
#define D_   512
#define H_   256
#define OUT_ 400
#define KP_IN 416   // inputs K=400 padded to multiple of 32

typedef __attribute__((ext_vector_type(16))) __bf16         v16bf;
typedef __attribute__((ext_vector_type(16))) unsigned short v16us;
typedef __attribute__((ext_vector_type(8)))  float          v8f;

__device__ __forceinline__ v8f wmma_bf16(v16bf a, v16bf b, v8f c) {
  // (neg_a, A, neg_b, B, c_mod, C, reuse_a, reuse_b)
  return __builtin_amdgcn_wmma_f32_16x16x32_bf16(false, a, false, b, (short)0, c,
                                                 false, false);
}

__device__ __forceinline__ v16bf bitcast_bf(v16us u) {
  union { v16us u; v16bf b; } t; t.u = u; return t.b;
}

// A fragment: 16x32 bf16 tile from fp32 row-major A (lda), unguarded.
// ISA layout: lane<16 -> kb=0, lane>=16 -> kb=8; elems 0..7 -> K=kb+e,
// elems 8..15 -> K=kb+16+(e-8).  => 4x float4 loads per lane.
__device__ __forceinline__ v16bf load_a_frag(const float* A, int lda, int row0,
                                             int k0) {
  int lane = threadIdx.x & 31;
  int m    = lane & 15;
  int kb   = (lane < 16) ? 0 : 8;
  const float* row = A + (size_t)(row0 + m) * lda + k0 + kb;
  float4 a0 = *(const float4*)(row);
  float4 a1 = *(const float4*)(row + 4);
  float4 a2 = *(const float4*)(row + 16);
  float4 a3 = *(const float4*)(row + 20);
  v16bf v;
  v[0]  = (__bf16)a0.x; v[1]  = (__bf16)a0.y; v[2]  = (__bf16)a0.z; v[3]  = (__bf16)a0.w;
  v[4]  = (__bf16)a1.x; v[5]  = (__bf16)a1.y; v[6]  = (__bf16)a1.z; v[7]  = (__bf16)a1.w;
  v[8]  = (__bf16)a2.x; v[9]  = (__bf16)a2.y; v[10] = (__bf16)a2.z; v[11] = (__bf16)a2.w;
  v[12] = (__bf16)a3.x; v[13] = (__bf16)a3.y; v[14] = (__bf16)a3.z; v[15] = (__bf16)a3.w;
  return v;
}

// B fragment: 32x16 bf16 tile from pre-transposed bf16 BT[N][Kp], unguarded.
// lane%16 = N column, kb = (lane<16?0:8); elems 0..7 -> K=k0+kb+e, elems 8..15 -> K=k0+kb+16+(e-8).
__device__ __forceinline__ v16bf load_b_frag(const unsigned short* BT, int Kp,
                                             int n0, int k0) {
  int lane = threadIdx.x & 31;
  int n    = n0 + (lane & 15);
  int base = (lane < 16) ? 0 : 8;
  const unsigned short* p = BT + (size_t)n * Kp + k0 + base;
  union { uint4 q[2]; v16us v; } u;
  u.q[0] = *(const uint4*)(p);
  u.q[1] = *(const uint4*)(p + 16);
  return bitcast_bf(u.v);
}

// ---------------- kernels ----------------

// src[K][N] fp32 -> dst[N][Kp] bf16, zero pad K..Kp-1
__global__ void transpose_bf16(const float* __restrict__ src,
                               unsigned short* __restrict__ dst,
                               int K, int N, int Kp) {
  int idx = blockIdx.x * blockDim.x + threadIdx.x;
  if (idx >= N * Kp) return;
  int n = idx / Kp, k = idx % Kp;
  float f = (k < K) ? src[(size_t)k * N + n] : 0.0f;
  dst[idx] = __builtin_bit_cast(unsigned short, (__bf16)f);
}

// src[M][K] -> dst[M][Kp] zero padded
__global__ void pad_copy(const float* __restrict__ src, float* __restrict__ dst,
                         int K, int Kp, int M) {
  int idx = blockIdx.x * blockDim.x + threadIdx.x;
  if (idx >= M * Kp) return;
  int m = idx / Kp, k = idx % Kp;
  dst[idx] = (k < K) ? src[(size_t)m * K + k] : 0.0f;
}

// C[M=128][N] = act(A[128][Kp](lda) @ B + bias), one wave per 16x16 tile.
// Kp must be a multiple of 32; pad regions of BT are zero.
__global__ void wmma_gemm(const float* __restrict__ A, int lda,
                          const unsigned short* __restrict__ BT,
                          const float* __restrict__ bias,
                          float* __restrict__ C, int ldc, int N, int Kp, int act) {
  int m0 = blockIdx.x * 16;
  int n0 = blockIdx.y * 16;
  v8f c = {};
  for (int k0 = 0; k0 < Kp; k0 += 32) {
    v16bf a = load_a_frag(A, lda, m0, k0);
    v16bf b = load_b_frag(BT, Kp, n0, k0);
    c = wmma_bf16(a, b, c);
  }
  int lane = threadIdx.x & 31;
  int n    = n0 + (lane & 15);
  int mo   = (lane < 16) ? 0 : 8;
  float bv = bias ? bias[n] : 0.0f;
#pragma unroll
  for (int r = 0; r < 8; ++r) {
    float v = c[r] + bv;
    if (act == 1) v = fmaxf(v, 0.0f);
    C[(size_t)(m0 + mo + r) * ldc + n] = v;
  }
}

// score[b,t] = v . tanh(q[b] + memory[b,t,:] @ Wk)   (fused, bf16 WMMA, f32 acc)
__global__ void attn_score_kernel(const float* __restrict__ mem,
                                  const unsigned short* __restrict__ WkT,
                                  const float* __restrict__ q,
                                  const float* __restrict__ v_attn,
                                  float* __restrict__ score) {
  int wave = threadIdx.x >> 5;
  int lane = threadIdx.x & 31;
  int b    = blockIdx.y;
  int t0   = blockIdx.x * 64 + wave * 16;
  const float* memb = mem + (size_t)b * T_ * D_;

  v16bf afr[16];                     // whole 16x512 row tile resident in VGPRs
#pragma unroll
  for (int kt = 0; kt < 16; ++kt)
    afr[kt] = load_a_frag(memb, D_, t0, kt * 32);

  float acc[8];
#pragma unroll
  for (int r = 0; r < 8; ++r) acc[r] = 0.0f;

  for (int nt = 0; nt < A_ / 16; ++nt) {
    int n0 = nt * 16;
    v8f c = {};
#pragma unroll
    for (int kt = 0; kt < 16; ++kt) {
      v16bf bfrag = load_b_frag(WkT, D_, n0, kt * 32);
      c = wmma_bf16(afr[kt], bfrag, c);
    }
    float qv = q[(size_t)b * A_ + n0 + (lane & 15)];
    float vv = v_attn[n0 + (lane & 15)];
#pragma unroll
    for (int r = 0; r < 8; ++r)
      acc[r] += tanhf(c[r] + qv) * vv;
  }
  // reduce over the 16 lanes sharing each row (xor 1,2,4,8 stays in half)
#pragma unroll
  for (int r = 0; r < 8; ++r) {
    float v = acc[r];
    v += __shfl_xor(v, 1, 32);
    v += __shfl_xor(v, 2, 32);
    v += __shfl_xor(v, 4, 32);
    v += __shfl_xor(v, 8, 32);
    if ((lane & 15) == 0) {
      int row = t0 + r + ((lane >> 4) << 3);
      score[(size_t)b * T_ + row] = v;
    }
  }
}

// out[b, off+d] = sum_t w[b,t] * mem[b,t,d]   (blockDim = D_; weights in LDS)
__global__ void ctx_kernel(const float* __restrict__ w,
                           const float* __restrict__ mem,
                           float* __restrict__ out, int ldo, int off) {
  __shared__ float sw[T_];
  int b = blockIdx.x, d = threadIdx.x;
  for (int t = d; t < T_; t += blockDim.x)
    sw[t] = w[(size_t)b * T_ + t];
  __syncthreads();
  const float* mb = mem + (size_t)b * T_ * D_ + d;
  float acc = 0.0f;
  for (int t = 0; t < T_; ++t)
    acc = fmaf(sw[t], mb[(size_t)t * D_], acc);
  out[(size_t)b * ldo + off + d] = acc;
}

// masked softmax over T, in place; mask is bool (1 byte)
__global__ void softmax_mask(float* __restrict__ score,
                             const unsigned char* __restrict__ mask) {
  __shared__ float red[256];
  int b = blockIdx.x, tid = threadIdx.x;
  float* s = score + (size_t)b * T_;
  const unsigned char* m = mask + (size_t)b * T_;
  float mx = -1e30f;
  for (int t = tid; t < T_; t += 256)
    mx = fmaxf(mx, m[t] ? s[t] : -1e9f);
  red[tid] = mx; __syncthreads();
  for (int st = 128; st > 0; st >>= 1) {
    if (tid < st) red[tid] = fmaxf(red[tid], red[tid + st]);
    __syncthreads();
  }
  mx = red[0]; __syncthreads();
  float sum = 0.0f;
  for (int t = tid; t < T_; t += 256)
    sum += expf((m[t] ? s[t] : -1e9f) - mx);
  red[tid] = sum; __syncthreads();
  for (int st = 128; st > 0; st >>= 1) {
    if (tid < st) red[tid] += red[tid + st];
    __syncthreads();
  }
  float inv = 1.0f / red[0];
  for (int t = tid; t < T_; t += 256)
    s[t] = expf((m[t] ? s[t] : -1e9f) - mx) * inv;
}

// Keras GRU (reset_after) gate math; writes out[b, off+j] with row stride ldo
__global__ void gru_combine(const float* __restrict__ gi,
                            const float* __restrict__ gr,
                            const float* __restrict__ h,
                            float* __restrict__ out, int n, int ldo, int off) {
  int idx = blockIdx.x * blockDim.x + threadIdx.x;
  if (idx >= B_ * n) return;
  int b = idx / n, j = idx % n;
  const float* gib = gi + (size_t)b * 3 * n;
  const float* grb = gr + (size_t)b * 3 * n;
  float z = 1.0f / (1.0f + expf(-(gib[j] + grb[j])));
  float r = 1.0f / (1.0f + expf(-(gib[n + j] + grb[n + j])));
  float c = tanhf(gib[2 * n + j] + r * grb[2 * n + j]);
  float hv = h[(size_t)b * n + j];
  out[(size_t)b * ldo + off + j] = z * hv + (1.0f - z) * c;
}

// ---------------- host ----------------

extern "C" void kernel_launch(void* const* d_in, const int* in_sizes, int n_in,
                              void* d_out, int out_size, void* d_ws, size_t ws_size,
                              hipStream_t stream) {
  const float* inputs          = (const float*)d_in[0];
  const float* prev_attn_h     = (const float*)d_in[1];
  const float* prev_dec_h1     = (const float*)d_in[2];
  const float* prev_dec_h2     = (const float*)d_in[3];
  const float* prev_alignments = (const float*)d_in[4];
  const float* memory          = (const float*)d_in[5];
  const unsigned char* memmask = (const unsigned char*)d_in[6];
  const float* Wp1 = (const float*)d_in[7],  *bp1 = (const float*)d_in[8];
  const float* Wp2 = (const float*)d_in[9],  *bp2 = (const float*)d_in[10];
  const float* Wq  = (const float*)d_in[11], *Wk  = (const float*)d_in[12];
  const float* v_attn = (const float*)d_in[13];
  const float* Wa  = (const float*)d_in[14], *ba  = (const float*)d_in[15];
  const float* Wg  = (const float*)d_in[16], *Ug  = (const float*)d_in[17];
  const float* bg_i = (const float*)d_in[18], *bg_r = (const float*)d_in[19];
  const float* Wd1 = (const float*)d_in[20], *Ud1 = (const float*)d_in[21];
  const float* bd1_i = (const float*)d_in[22], *bd1_r = (const float*)d_in[23];
  const float* Wd2 = (const float*)d_in[24], *Ud2 = (const float*)d_in[25];
  const float* bd2_i = (const float*)d_in[26], *bd2_r = (const float*)d_in[27];
  const float* Wo  = (const float*)d_in[28], *bo  = (const float*)d_in[29];
  float* out = (float*)d_out;

  char* ws = (char*)d_ws;
  size_t off = 0;
  auto alloc = [&](size_t bytes) -> void* {
    void* p = ws + off;
    off += (bytes + 255) & ~(size_t)255;
    return p;
  };
  // bf16 transposed weights (row length = Kp, multiple of 32)
  unsigned short* WkT  = (unsigned short*)alloc((size_t)A_ * D_ * 2);
  unsigned short* Wp1T = (unsigned short*)alloc((size_t)E_ * KP_IN * 2);
  unsigned short* Wp2T = (unsigned short*)alloc((size_t)H_ * E_ * 2);
  unsigned short* WqT  = (unsigned short*)alloc((size_t)A_ * E_ * 2);
  unsigned short* WaT  = (unsigned short*)alloc((size_t)E_ * D_ * 2);
  unsigned short* WgT  = (unsigned short*)alloc((size_t)(3 * E_) * (H_ + E_) * 2);
  unsigned short* UgT  = (unsigned short*)alloc((size_t)(3 * E_) * E_ * 2);
  unsigned short* Wd1T = (unsigned short*)alloc((size_t)(3 * H_) * (E_ + D_) * 2);
  unsigned short* Ud1T = (unsigned short*)alloc((size_t)(3 * H_) * H_ * 2);
  unsigned short* Wd2T = (unsigned short*)alloc((size_t)(3 * H_) * H_ * 2);
  unsigned short* Ud2T = (unsigned short*)alloc((size_t)(3 * H_) * H_ * 2);
  unsigned short* WoT  = (unsigned short*)alloc((size_t)OUT_ * H_ * 2);
  // fp32 activations
  float* inputs_p = (float*)alloc((size_t)B_ * KP_IN * 4);
  float* prev_ctx = (float*)alloc((size_t)B_ * D_ * 4);
  float* attn_in  = (float*)alloc((size_t)B_ * (H_ + E_) * 4);
  float* pre1     = (float*)alloc((size_t)B_ * E_ * 4);
  float* gi_g     = (float*)alloc((size_t)B_ * 3 * E_ * 4);
  float* gr_g     = (float*)alloc((size_t)B_ * 3 * E_ * 4);
  float* dec_in   = (float*)alloc((size_t)B_ * (E_ + D_) * 4);
  float* qbuf     = (float*)alloc((size_t)B_ * A_ * 4);
  float* score    = (float*)alloc((size_t)B_ * T_ * 4);
  float* gi_d1    = (float*)alloc((size_t)B_ * 3 * H_ * 4);
  float* gr_d1    = (float*)alloc((size_t)B_ * 3 * H_ * 4);
  float* gi_d2    = (float*)alloc((size_t)B_ * 3 * H_ * 4);
  float* gr_d2    = (float*)alloc((size_t)B_ * 3 * H_ * 4);
  float* h1       = (float*)alloc((size_t)B_ * H_ * 4);
  float* h2       = (float*)alloc((size_t)B_ * H_ * 4);

  auto tconv = [&](const float* src, unsigned short* dst, int K, int N, int Kp) {
    int total = N * Kp;
    transpose_bf16<<<dim3((total + 255) / 256), dim3(256), 0, stream>>>(
        src, dst, K, N, Kp);
  };
  auto gemm = [&](const float* A, int lda, const unsigned short* BT,
                  const float* bias, float* C, int ldc, int N, int Kp, int act) {
    wmma_gemm<<<dim3(B_ / 16, N / 16), dim3(32), 0, stream>>>(A, lda, BT, bias, C,
                                                              ldc, N, Kp, act);
  };

  // 0) weight preprocessing (transpose + bf16 + pad) and input padding
  tconv(Wk,  WkT,  D_, A_, D_);
  tconv(Wp1, Wp1T, OUT_, E_, KP_IN);
  tconv(Wp2, Wp2T, E_, H_, E_);
  tconv(Wq,  WqT,  E_, A_, E_);
  tconv(Wa,  WaT,  D_, E_, D_);
  tconv(Wg,  WgT,  H_ + E_, 3 * E_, H_ + E_);
  tconv(Ug,  UgT,  E_, 3 * E_, E_);
  tconv(Wd1, Wd1T, E_ + D_, 3 * H_, E_ + D_);
  tconv(Ud1, Ud1T, H_, 3 * H_, H_);
  tconv(Wd2, Wd2T, H_, 3 * H_, H_);
  tconv(Ud2, Ud2T, H_, 3 * H_, H_);
  tconv(Wo,  WoT,  H_, OUT_, H_);
  {
    int total = B_ * KP_IN;
    pad_copy<<<dim3((total + 255) / 256), dim3(256), 0, stream>>>(
        inputs, inputs_p, OUT_, KP_IN, B_);
  }

  // 1) previous context + attention_layer
  ctx_kernel<<<dim3(B_), dim3(D_), 0, stream>>>(prev_alignments, memory,
                                                prev_ctx, D_, 0);
  gemm(prev_ctx, D_, WaT, ba, attn_in + H_, H_ + E_, E_, D_, 0);
  // 2) prenet
  gemm(inputs_p, KP_IN, Wp1T, bp1, pre1, E_, E_, KP_IN, 1);
  gemm(pre1, E_, Wp2T, bp2, attn_in, H_ + E_, H_, E_, 1);
  // 3) attention GRU -> attn_h stored in dec_in[:, 0:E]
  gemm(attn_in, H_ + E_, WgT, bg_i, gi_g, 3 * E_, 3 * E_, H_ + E_, 0);
  gemm(prev_attn_h, E_, UgT, bg_r, gr_g, 3 * E_, 3 * E_, E_, 0);
  {
    int total = B_ * E_;
    gru_combine<<<dim3((total + 255) / 256), dim3(256), 0, stream>>>(
        gi_g, gr_g, prev_attn_h, dec_in, E_, E_ + D_, 0);
  }
  // 4) Bahdanau attention
  gemm(dec_in, E_ + D_, WqT, nullptr, qbuf, A_, A_, E_, 0);
  attn_score_kernel<<<dim3(T_ / 64, B_), dim3(128), 0, stream>>>(
      memory, WkT, qbuf, v_attn, score);
  softmax_mask<<<dim3(B_), dim3(256), 0, stream>>>(score, memmask);
  ctx_kernel<<<dim3(B_), dim3(D_), 0, stream>>>(score, memory, dec_in,
                                                E_ + D_, E_);
  // 5) decoder GRU stack
  gemm(dec_in, E_ + D_, Wd1T, bd1_i, gi_d1, 3 * H_, 3 * H_, E_ + D_, 0);
  gemm(prev_dec_h1, H_, Ud1T, bd1_r, gr_d1, 3 * H_, 3 * H_, H_, 0);
  {
    int total = B_ * H_;
    gru_combine<<<dim3((total + 255) / 256), dim3(256), 0, stream>>>(
        gi_d1, gr_d1, prev_dec_h1, h1, H_, H_, 0);
  }
  gemm(h1, H_, Wd2T, bd2_i, gi_d2, 3 * H_, 3 * H_, H_, 0);
  gemm(prev_dec_h2, H_, Ud2T, bd2_r, gr_d2, 3 * H_, 3 * H_, H_, 0);
  {
    int total = B_ * H_;
    gru_combine<<<dim3((total + 255) / 256), dim3(256), 0, stream>>>(
        gi_d2, gr_d2, prev_dec_h2, h2, H_, H_, 0);
  }
  // 6) output projection straight into d_out
  gemm(h2, H_, WoT, bo, out, OUT_, OUT_, H_, 0);
  (void)in_sizes; (void)n_in; (void)out_size; (void)ws_size;
}
